// CrossAttention_18863496364354
// MI455X (gfx1250) — matrix operand order repaired and checked by reading, not compiled, not run on hardware
//
#include <hip/hip_runtime.h>
#include <stdint.h>

// ---------------------------------------------------------------------------
// CDNA5 (gfx1250) wave32 WMMA types
// ---------------------------------------------------------------------------
typedef __attribute__((ext_vector_type(16))) __bf16          bf16x16;
typedef __attribute__((ext_vector_type(8)))  float           f32x8;
typedef __attribute__((ext_vector_type(8)))  unsigned short  u16x8;

union FragAB {
    bf16x16 bf;
    u16x8   u[2];
};

static __device__ __forceinline__ unsigned short f2bf(float f) {
    unsigned int u = __float_as_uint(f);
    u += 0x7FFFu + ((u >> 16) & 1u);   // round-to-nearest-even
    return (unsigned short)(u >> 16);
}

static __device__ __forceinline__ f32x8 zero8() {
    f32x8 z = {0.f, 0.f, 0.f, 0.f, 0.f, 0.f, 0.f, 0.f};
    return z;
}

// gfx1250 async copy: global -> LDS, 16 bytes per lane, tracked by ASYNCcnt.
static __device__ __forceinline__ void async_copy_b128(const void* gsrc, void* lds_dst) {
    unsigned           ldsoff = (unsigned)(size_t)lds_dst;     // low 32 bits = LDS offset
    unsigned long long gaddr  = (unsigned long long)(size_t)gsrc;
    asm volatile("global_load_async_to_lds_b128 %0, %1, off"
                 :: "v"(ldsoff), "v"(gaddr) : "memory");
}
static __device__ __forceinline__ void wait_asynccnt0() {
    asm volatile("s_wait_asynccnt 0x0" ::: "memory");
}

// ---------------------------------------------------------------------------
// Tiled GEMM:  C[Rows x Cols] = A[Rows x K] * W[K x Cols]   (bf16 WMMA, f32 acc)
//   A_BF16  : A is bf16 workspace -> staged to LDS via global_load_async_to_lds
//   OUT_F32 : write fp32 + bias  vs  bf16 * outScale
// block = 256 threads = 8 waves; block tile 128x128; wave tile 32x64;
// K-step 32; double-buffered LDS (one barrier per K-step).
// ---------------------------------------------------------------------------
template <bool A_BF16, bool OUT_F32>
__global__ __launch_bounds__(256)
void gemm_bf16_wmma(const void* __restrict__ Aptr,
                    const float* __restrict__ W,
                    const float* __restrict__ bias,
                    void* __restrict__ Cptr,
                    int Rows, int K, int Cols, float outScale)
{
    __shared__ unsigned short As[2][128 * 32];   // [row][k]   row-major
    __shared__ unsigned short Bt[2][128 * 32];   // [col][k]   k-contiguous (transposed)

    const int tid  = threadIdx.x;
    const int lane = tid & 31;
    const int wave = tid >> 5;
    const int wm   = wave >> 1;          // 0..3  -> 32-row strip
    const int wn   = wave & 1;           // 0..1  -> 64-col strip
    const int r0   = blockIdx.y * 128;
    const int n0   = blockIdx.x * 128;
    const int l15  = lane & 15;
    const int b8   = (lane >= 16) ? 8 : 0;    // A-frag K offset, upper half-wave
    const int koff = (lane >= 16) ? 8 : 0;    // B-frag K offset, upper half-wave

    f32x8 acc[2][4];
#pragma unroll
    for (int i = 0; i < 2; ++i)
#pragma unroll
        for (int j = 0; j < 4; ++j) acc[i][j] = zero8();

    // ---- tile staging (global -> LDS bf16) ---------------------------------
    auto stage = [&](int buf, int k0) {
        unsigned short* Asb = &As[buf][0];
        unsigned short* Btb = &Bt[buf][0];
        if (A_BF16) {
            const unsigned short* A = (const unsigned short*)Aptr;
#pragma unroll
            for (int i = 0; i < 2; ++i) {
                int idx = tid + i * 256;
                int row = idx >> 2, c = idx & 3;
                async_copy_b128(A + (size_t)(r0 + row) * K + k0 + c * 8,
                                Asb + row * 32 + c * 8);
            }
        } else {
            const float* A = (const float*)Aptr;
#pragma unroll
            for (int i = 0; i < 4; ++i) {
                int idx = tid + i * 256;
                int row = idx >> 3, c = idx & 7;
                float4 f = *(const float4*)(A + (size_t)(r0 + row) * K + k0 + c * 4);
                unsigned int lo = (unsigned int)f2bf(f.x) | ((unsigned int)f2bf(f.y) << 16);
                unsigned int hi = (unsigned int)f2bf(f.z) | ((unsigned int)f2bf(f.w) << 16);
                unsigned int* p = (unsigned int*)(Asb + row * 32 + c * 4);
                p[0] = lo; p[1] = hi;
            }
        }
#pragma unroll
        for (int i = 0; i < 4; ++i) {
            int idx = tid + i * 256;
            int kk = idx >> 5, nq = idx & 31;
            float4 f = *(const float4*)(W + (size_t)(k0 + kk) * Cols + n0 + nq * 4);
            Btb[(nq * 4 + 0) * 32 + kk] = f2bf(f.x);
            Btb[(nq * 4 + 1) * 32 + kk] = f2bf(f.y);
            Btb[(nq * 4 + 2) * 32 + kk] = f2bf(f.z);
            Btb[(nq * 4 + 3) * 32 + kk] = f2bf(f.w);
        }
    };

    const int KT = K >> 5;
    stage(0, 0);

    for (int kt = 0; kt < KT; ++kt) {
        const int cur = kt & 1;
        if (A_BF16) wait_asynccnt0();        // our async writes to LDS landed
        __syncthreads();                     // tile `kt` visible to all waves
        if (kt + 1 < KT) stage(cur ^ 1, (kt + 1) << 5);   // overlap next tile

        // ---- gather fragments (two b128 LDS reads each) and WMMA ----
        const unsigned short* Asb = &As[cur][0];
        const unsigned short* Btb = &Bt[cur][0];
        FragAB fa[2], fb[4];
#pragma unroll
        for (int im = 0; im < 2; ++im) {
            const unsigned short* p = Asb + (wm * 32 + im * 16 + l15) * 32 + b8;
            fa[im].u[0] = *(const u16x8*)p;
            fa[im].u[1] = *(const u16x8*)(p + 16);
        }
#pragma unroll
        for (int in = 0; in < 4; ++in) {
            const unsigned short* p = Btb + (wn * 64 + in * 16 + l15) * 32 + koff;
            fb[in].u[0] = *(const u16x8*)p;
            fb[in].u[1] = *(const u16x8*)(p + 16);
        }
#pragma unroll
        for (int im = 0; im < 2; ++im)
#pragma unroll
            for (int in = 0; in < 4; ++in)
                acc[im][in] = __builtin_amdgcn_wmma_f32_16x16x32_bf16(
                    false, fa[im].bf, false, fb[in].bf, (short)0, acc[im][in],
                    false, false);
    }

    // ---- epilogue: C layout = VGPR r -> row r (lanes 0-15) / r+8 (lanes 16-31)
    const int rsel = (lane >= 16) ? 8 : 0;
#pragma unroll
    for (int im = 0; im < 2; ++im) {
        int rowg = r0 + wm * 32 + im * 16 + rsel;
#pragma unroll
        for (int in = 0; in < 4; ++in) {
            int colg = n0 + wn * 64 + in * 16 + l15;
#pragma unroll
            for (int r = 0; r < 8; ++r) {
                float val = acc[im][in][r];
                if (OUT_F32) {
                    float* Co = (float*)Cptr;
                    Co[(size_t)(rowg + r) * Cols + colg] = val + bias[colg];
                } else {
                    unsigned short* Co = (unsigned short*)Cptr;
                    Co[(size_t)(rowg + r) * Cols + colg] = f2bf(val * outScale);
                }
            }
        }
    }
}

// ---------------------------------------------------------------------------
// Flash-style attention over bf16 q/k/v (q pre-scaled by 1/sqrt(d)).
// grid = (N/64, HEADS, B), block = 128 (4 waves); each wave owns 16 query rows.
// S = Q K^T (2 chained WMMA over d=64), online softmax in C-layout,
// P -> LDS (bf16, A-layout re-read), O += P V (WMMA, k = 32 keys per chunk).
// Double-buffered V tile: one barrier per chunk; L2 prefetch 2 chunks ahead.
// ---------------------------------------------------------------------------
__global__ __launch_bounds__(128)
void attn_fwd_wmma(const unsigned short* __restrict__ q,
                   const unsigned short* __restrict__ kmat,
                   const unsigned short* __restrict__ vmat,
                   unsigned short* __restrict__ o)
{
    constexpr int Nq = 1024, Mk = 1024, INNER = 1024;
    __shared__ unsigned short Vt[2][64 * 32];     // [d][m]  m-contiguous
    __shared__ unsigned short Pl[4 * 16 * 32];    // per-wave P tile [row][m]

    const int tid  = threadIdx.x;
    const int lane = tid & 31;
    const int wave = tid >> 5;
    const int bb   = blockIdx.z;
    const int h    = blockIdx.y;
    const int n0w  = blockIdx.x * 64 + wave * 16;
    const int l15  = lane & 15;
    const int b8   = (lane >= 16) ? 8 : 0;
    const int koff = (lane >= 16) ? 8 : 0;

    // Q A-fragments straight from global (head dim contiguous): two b128 each
    FragAB qa[2];
    {
        size_t base = ((size_t)(bb * Nq + n0w + l15)) * INNER + h * 64 + b8;
        qa[0].u[0] = *(const u16x8*)(q + base);
        qa[0].u[1] = *(const u16x8*)(q + base + 16);
        qa[1].u[0] = *(const u16x8*)(q + base + 32);
        qa[1].u[1] = *(const u16x8*)(q + base + 48);
    }

    f32x8 oacc[4];
#pragma unroll
    for (int c = 0; c < 4; ++c) oacc[c] = zero8();
    float rmax[8], rsum[8];
#pragma unroll
    for (int r = 0; r < 8; ++r) { rmax[r] = -3.0e38f; rsum[r] = 0.0f; }

    unsigned short* Pw = Pl + wave * 16 * 32;

    // cooperative V chunk load, transposed into Vt[buf][d][m]
    auto stageV = [&](int buf, int m0c) {
        unsigned short* Vb = &Vt[buf][0];
#pragma unroll
        for (int i = 0; i < 16; ++i) {
            int idx = i * 128 + tid;
            int d = idx & 63, mm = idx >> 6;
            Vb[d * 32 + mm] =
                vmat[((size_t)(bb * Mk + m0c + mm)) * INNER + h * 64 + d];
        }
    };

    const int NC = Mk / 32;
    stageV(0, 0);

    for (int t = 0; t < NC; ++t) {
        const int cur = t & 1;
        const int m0c = t * 32;
        __syncthreads();                               // Vt[cur] ready for all
        if (t + 1 < NC) stageV(cur ^ 1, m0c + 32);     // overlap next chunk
        if (t + 2 < NC) {                              // prime L2 two chunks out
            size_t prow = ((size_t)(bb * Mk + m0c + 64 + lane)) * INNER
                          + h * 64 + (wave << 4);
            __builtin_prefetch(vmat + prow, 0, 1);     // -> global_prefetch_b8
            __builtin_prefetch(kmat + prow, 0, 1);
        }

        // S = Q K^T for two 16-column blocks (K B-frags direct from global)
        f32x8 s[2];
#pragma unroll
        for (int mb = 0; mb < 2; ++mb) {
            size_t kbase =
                ((size_t)(bb * Mk + m0c + mb * 16 + l15)) * INNER + h * 64 + koff;
            FragAB kb0, kb1;
            kb0.u[0] = *(const u16x8*)(kmat + kbase);
            kb0.u[1] = *(const u16x8*)(kmat + kbase + 16);
            kb1.u[0] = *(const u16x8*)(kmat + kbase + 32);
            kb1.u[1] = *(const u16x8*)(kmat + kbase + 48);
            f32x8 a = zero8();
            a = __builtin_amdgcn_wmma_f32_16x16x32_bf16(
                    false, qa[0].bf, false, kb0.bf, (short)0, a, false, false);
            a = __builtin_amdgcn_wmma_f32_16x16x32_bf16(
                    false, qa[1].bf, false, kb1.bf, (short)0, a, false, false);
            s[mb] = a;
        }

        // online softmax in C-layout: row r lives in accumulator element r,
        // spread over 16 lanes of each half-wave -> xor-shuffle reductions
#pragma unroll
        for (int r = 0; r < 8; ++r) {
            float cm = fmaxf(s[0][r], s[1][r]);
            cm = fmaxf(cm, __shfl_xor(cm, 1));
            cm = fmaxf(cm, __shfl_xor(cm, 2));
            cm = fmaxf(cm, __shfl_xor(cm, 4));
            cm = fmaxf(cm, __shfl_xor(cm, 8));
            float nm = fmaxf(rmax[r], cm);
            float sc = __expf(rmax[r] - nm);
            float p0 = __expf(s[0][r] - nm);
            float p1 = __expf(s[1][r] - nm);
            float ps = p0 + p1;
            ps += __shfl_xor(ps, 1);
            ps += __shfl_xor(ps, 2);
            ps += __shfl_xor(ps, 4);
            ps += __shfl_xor(ps, 8);
            rsum[r] = rsum[r] * sc + ps;
            rmax[r] = nm;
#pragma unroll
            for (int c = 0; c < 4; ++c) oacc[c][r] *= sc;
            int rr = r + b8;
            Pw[rr * 32 + l15]      = f2bf(p0);        // wave-private P tile;
            Pw[rr * 32 + l15 + 16] = f2bf(p1);        // same-wave DS is in-order
        }

        // O += P @ V  (A-frag of P from LDS, B-frag of V^T from LDS)
        FragAB pa;
        {
            const unsigned short* p = Pw + l15 * 32 + b8;
            pa.u[0] = *(const u16x8*)p;
            pa.u[1] = *(const u16x8*)(p + 16);
        }
        const unsigned short* Vb = &Vt[cur][0];
#pragma unroll
        for (int c = 0; c < 4; ++c) {
            FragAB vb;
            const unsigned short* p = Vb + (c * 16 + l15) * 32 + koff;
            vb.u[0] = *(const u16x8*)p;
            vb.u[1] = *(const u16x8*)(p + 16);
            oacc[c] = __builtin_amdgcn_wmma_f32_16x16x32_bf16(
                false, pa.bf, false, vb.bf, (short)0, oacc[c], false, false);
        }
    }

    // finalize: divide by softmax denominator, store bf16 (b, n, h*64+d)
#pragma unroll
    for (int r = 0; r < 8; ++r) {
        float inv = 1.0f / rsum[r];
        int n = n0w + r + b8;
#pragma unroll
        for (int c = 0; c < 4; ++c) {
            o[((size_t)(bb * Nq + n)) * INNER + h * 64 + c * 16 + l15] =
                f2bf(oacc[c][r] * inv);
        }
    }
}

// ---------------------------------------------------------------------------
// Host-side launcher
// inputs: x, context, Wq, Wk, Wv, Wo, bo (all fp32); output fp32 [B,N,D]
// ---------------------------------------------------------------------------
extern "C" void kernel_launch(void* const* d_in, const int* in_sizes, int n_in,
                              void* d_out, int out_size, void* d_ws, size_t ws_size,
                              hipStream_t stream)
{
    (void)in_sizes; (void)n_in; (void)out_size; (void)ws_size;
    const float* x   = (const float*)d_in[0];
    const float* ctx = (const float*)d_in[1];
    const float* Wq  = (const float*)d_in[2];
    const float* Wk  = (const float*)d_in[3];
    const float* Wv  = (const float*)d_in[4];
    const float* Wo  = (const float*)d_in[5];
    const float* bo  = (const float*)d_in[6];
    float* out = (float*)d_out;

    const int Bb = 8, Nn = 1024, Mm = 1024, Dm = 1024, In = 1024;
    const float SCALE = 0.125f;   // 64^-0.5

    size_t e = (size_t)Bb * Nn * In;            // 8M elements each
    unsigned short* qw = (unsigned short*)d_ws; // bf16 workspace: q,k,v,o
    unsigned short* kw = qw + e;
    unsigned short* vw = kw + e;
    unsigned short* ow = vw + e;

    dim3 gblk(256), ggrid(In / 128, (Bb * Nn) / 128);
    gemm_bf16_wmma<false, false><<<ggrid, gblk, 0, stream>>>(
        x,   Wq, nullptr, qw, Bb * Nn, Dm, In, SCALE);
    gemm_bf16_wmma<false, false><<<ggrid, gblk, 0, stream>>>(
        ctx, Wk, nullptr, kw, Bb * Mm, Dm, In, 1.0f);
    gemm_bf16_wmma<false, false><<<ggrid, gblk, 0, stream>>>(
        ctx, Wv, nullptr, vw, Bb * Mm, Dm, In, 1.0f);

    dim3 ablk(128), agrid(Nn / 64, 16, Bb);
    attn_fwd_wmma<<<agrid, ablk, 0, stream>>>(qw, kw, vw, ow);

    dim3 ogrid(Dm / 128, (Bb * Nn) / 128);
    gemm_bf16_wmma<true, true><<<ogrid, gblk, 0, stream>>>(
        ow, Wo, bo, out, Bb * Nn, In, Dm, 1.0f);
}
